// LinearLayer_MatrixSparsity_9801115369812
// MI455X (gfx1250) — matrix operand order repaired and checked by reading, not compiled, not run on hardware
//
#include <hip/hip_runtime.h>

// ---------------------------------------------------------------------------
// LinearLayer_MatrixSparsity on MI455X (gfx1250)
//   out[8192,4096] = x[8192,4096] . Wnew[4096,4096]^T
// Plan:
//   1) prep_x : split f32 x into bf16 hi/lo pair (workspace)
//   2) prep_w : scatter 32 trainable 256x256 blocks over frozen W, split to
//               bf16 hi/lo pair (workspace)
//   3) gemm   : 128x128x32-tiled, 8-wave WG, double-buffered LDS fed by
//               GLOBAL_LOAD_ASYNC_TO_LDS_B128 (ASYNCcnt), 3-term bf16 WMMA
//               (hi*hi + hi*lo + lo*hi) with f32 accumulation
// Workspace requirement: 192 MiB (x pair 128 MiB + W pair 64 MiB).
// ---------------------------------------------------------------------------

typedef __attribute__((ext_vector_type(16))) __bf16       v16bf;
typedef __attribute__((ext_vector_type(8)))  float        v8f;
typedef __attribute__((ext_vector_type(4))) unsigned int  u32x4;
typedef __attribute__((ext_vector_type(2))) unsigned int  u32x2;
typedef __attribute__((ext_vector_type(4))) float         f32x4;

#define M_TOT 8192
#define N_TOT 4096
#define K_TOT 4096
#define BLKSZ 256
#define GO    16

#define TM 128
#define TN 128
#define TK 32
#define LDSK 40   // padded K-stride in LDS (80B rows, keeps 16B alignment)

// ---- CDNA5 async global->LDS path (guarded; falls back to reg staging) -----
#if defined(__has_builtin)
#if __has_builtin(__builtin_amdgcn_global_load_async_to_lds_b128)
#define HAVE_ASYNC_LDS 1
#endif
#endif

#ifdef HAVE_ASYNC_LDS
// Builtin's param 0 printed by clang as:
//   '__attribute__((__vector_size__(4 * sizeof(int)))) int __device__ *'
// i.e. pointer to 16B int vector in address_space(1) (HIP prints AS1 as
// __device__). Match it exactly; LDS side uses address_space(3).
typedef int i32x4 __attribute__((vector_size(16)));
typedef __attribute__((address_space(1))) i32x4* as1_i32x4p;
typedef __attribute__((address_space(3))) i32x4* as3_i32x4p;

__device__ __forceinline__ void async_copy_b128(const void* g, void* l) {
  __builtin_amdgcn_global_load_async_to_lds_b128(
      (as1_i32x4p)(i32x4*)(void*)g,
      (as3_i32x4p)(i32x4*)l,
      0 /*imm offset*/, 0 /*cpol*/);
}
#if __has_builtin(__builtin_amdgcn_s_wait_asynccnt)
#define WAIT_ASYNC0() __builtin_amdgcn_s_wait_asynccnt(0)
#else
#define WAIT_ASYNC0() asm volatile("s_wait_asynccnt 0x0" ::: "memory")
#endif
#endif

union P4 { __bf16 b[4]; u32x2 u; };

// ---- split one float into bf16 hi + bf16 lo --------------------------------
__device__ __forceinline__ void split_bf16(float f, __bf16& h, __bf16& l) {
  h = (__bf16)f;
  l = (__bf16)(f - (float)h);
}

// ---- prep_x: x f32 -> (x_hi, x_lo) bf16 ------------------------------------
__global__ void prep_x_kernel(const float* __restrict__ x,
                              __bf16* __restrict__ xh,
                              __bf16* __restrict__ xl) {
  size_t i = ((size_t)blockIdx.x * blockDim.x + threadIdx.x) * 4;
  f32x4 v = *(const f32x4*)(x + i);
  P4 h, l;
  split_bf16(v.x, h.b[0], l.b[0]);
  split_bf16(v.y, h.b[1], l.b[1]);
  split_bf16(v.z, h.b[2], l.b[2]);
  split_bf16(v.w, h.b[3], l.b[3]);
  *(u32x2*)(xh + i) = h.u;
  *(u32x2*)(xl + i) = l.u;
}

// ---- prep_w: scatter blocks + split to (w_hi, w_lo) bf16 -------------------
__global__ void prep_w_kernel(const float* __restrict__ w,
                              const float* __restrict__ sel,
                              const int*   __restrict__ idx,
                              __bf16* __restrict__ wh,
                              __bf16* __restrict__ wl) {
  __shared__ int map[GO * GO];
  int t = threadIdx.x;
  map[t] = -1;
  __syncthreads();
  if (t < 32) {
    int r = idx[2 * t];
    int c = idx[2 * t + 1];
    map[r * GO + c] = t;
  }
  __syncthreads();

  size_t e = ((size_t)blockIdx.x * BLKSZ + t) * 4;
  int n = (int)(e >> 12);       // row of W (output-feature)
  int k = (int)(e & 4095);      // col of W (input-feature)
  int b = map[((n >> 8) * GO) + (k >> 8)];
  f32x4 v;
  if (b >= 0) {
    const float* p = sel + (size_t)b * (BLKSZ * BLKSZ)
                         + (size_t)(n & 255) * BLKSZ + (k & 255);
    v = *(const f32x4*)p;
  } else {
    v = *(const f32x4*)(w + e);
  }
  P4 h, l;
  split_bf16(v.x, h.b[0], l.b[0]);
  split_bf16(v.y, h.b[1], l.b[1]);
  split_bf16(v.z, h.b[2], l.b[2]);
  split_bf16(v.w, h.b[3], l.b[3]);
  *(u32x2*)(wh + e) = h.u;
  *(u32x2*)(wl + e) = l.u;
}

// ---- WMMA fragment loads from LDS ------------------------------------------
// A (16x32 bf16): lanes 0-15 row=lr K=0..7,16..23 ; lanes 16-31 row=lr K=8..15,24..31
union FragU { v16bf v; u32x4 q[2]; };

__device__ __forceinline__ v16bf frag_a(const __bf16* tile, int row, int half) {
  FragU f;
  const __bf16* p = tile + row * LDSK + half * 8;
  f.q[0] = *(const u32x4*)p;
  f.q[1] = *(const u32x4*)(p + 16);
  return f.v;
}
// B (32x16 bf16): lanes 0-15 col=lr K=0..7,16..23 ; lanes 16-31 col=lr K=8..15,24..31
__device__ __forceinline__ v16bf frag_b(const __bf16* tile, int row, int half) {
  FragU f;
  const __bf16* p = tile + row * LDSK + half * 8;
  f.q[0] = *(const u32x4*)p;
  f.q[1] = *(const u32x4*)(p + 16);
  return f.v;
}

// ---- GEMM ------------------------------------------------------------------
__global__ __launch_bounds__(256)
void gemm_kernel(const __bf16* __restrict__ xh, const __bf16* __restrict__ xl,
                 const __bf16* __restrict__ wh, const __bf16* __restrict__ wl,
                 float* __restrict__ out) {
  __shared__ __attribute__((aligned(16))) __bf16 sxh[2][TM * LDSK];
  __shared__ __attribute__((aligned(16))) __bf16 sxl[2][TM * LDSK];
  __shared__ __attribute__((aligned(16))) __bf16 swh[2][TN * LDSK];
  __shared__ __attribute__((aligned(16))) __bf16 swl[2][TN * LDSK];

  const int t    = threadIdx.x;
  const int lane = t & 31;
  const int lr   = lane & 15;
  const int half = (lane >> 4) & 1;
  const int wave = t >> 5;     // 0..7
  const int wm   = wave & 1;   // 2 waves along M (64 rows each)
  const int wn   = wave >> 1;  // 4 waves along N (32 cols each)
  const int m0   = blockIdx.y * TM;
  const int n0   = blockIdx.x * TN;

  // global staging: each thread moves 2x16B per array per stage
  int rows[2], kcs[2];
#pragma unroll
  for (int i = 0; i < 2; i++) {
    int ch = t + i * 256;
    rows[i] = ch >> 2;          // 0..127
    kcs[i]  = (ch & 3) * 8;     // 0,8,16,24
  }

#ifdef HAVE_ASYNC_LDS
  // -- async copy engine path: global -> LDS directly, tracked by ASYNCcnt --
  auto issue_stage = [&](int k0, int buf) {
#pragma unroll
    for (int i = 0; i < 2; i++) {
      size_t xo = (size_t)(m0 + rows[i]) * K_TOT + k0 + kcs[i];
      size_t wo = (size_t)(n0 + rows[i]) * K_TOT + k0 + kcs[i];
      int off = rows[i] * LDSK + kcs[i];
      async_copy_b128(xh + xo, &sxh[buf][off]);
      async_copy_b128(xl + xo, &sxl[buf][off]);
      async_copy_b128(wh + wo, &swh[buf][off]);
      async_copy_b128(wl + wo, &swl[buf][off]);
    }
  };
#else
  u32x4 rxh[2], rxl[2], rwh[2], rwl[2];
  auto g_load = [&](int k0) {
#pragma unroll
    for (int i = 0; i < 2; i++) {
      size_t xo = (size_t)(m0 + rows[i]) * K_TOT + k0 + kcs[i];
      size_t wo = (size_t)(n0 + rows[i]) * K_TOT + k0 + kcs[i];
      rxh[i] = *(const u32x4*)(xh + xo);
      rxl[i] = *(const u32x4*)(xl + xo);
      rwh[i] = *(const u32x4*)(wh + wo);
      rwl[i] = *(const u32x4*)(wl + wo);
    }
  };
  auto l_store = [&](int buf) {
#pragma unroll
    for (int i = 0; i < 2; i++) {
      int off = rows[i] * LDSK + kcs[i];
      *(u32x4*)(&sxh[buf][off]) = rxh[i];
      *(u32x4*)(&sxl[buf][off]) = rxl[i];
      *(u32x4*)(&swh[buf][off]) = rwh[i];
      *(u32x4*)(&swl[buf][off]) = rwl[i];
    }
  };
#endif

  v8f acc[4][2];
#pragma unroll
  for (int mi = 0; mi < 4; mi++)
#pragma unroll
    for (int nj = 0; nj < 2; nj++) {
      v8f z = {};
      acc[mi][nj] = z;
    }

#ifdef HAVE_ASYNC_LDS
  issue_stage(0, 0);
  WAIT_ASYNC0();
  __syncthreads();
#else
  g_load(0);
  l_store(0);
  __syncthreads();
#endif

  const int KT = K_TOT / TK;  // 128
  for (int kt = 0; kt < KT; kt++) {
    const int cur = kt & 1, nxt = cur ^ 1;

#ifdef HAVE_ASYNC_LDS
    if (kt + 1 < KT) issue_stage((kt + 1) * TK, nxt);
#else
    if (kt + 1 < KT) g_load((kt + 1) * TK);
#endif

    v16bf ah[4], al[4], bh[2], bl[2];
#pragma unroll
    for (int mi = 0; mi < 4; mi++) {
      int row = wm * 64 + mi * 16 + lr;
      ah[mi] = frag_a(&sxh[cur][0], row, half);
      al[mi] = frag_a(&sxl[cur][0], row, half);
    }
#pragma unroll
    for (int nj = 0; nj < 2; nj++) {
      int row = wn * 32 + nj * 16 + lr;
      bh[nj] = frag_b(&swh[cur][0], row, half);
      bl[nj] = frag_b(&swl[cur][0], row, half);
    }

#pragma unroll
    for (int mi = 0; mi < 4; mi++) {
#pragma unroll
      for (int nj = 0; nj < 2; nj++) {
        acc[mi][nj] = __builtin_amdgcn_wmma_f32_16x16x32_bf16(
            false, ah[mi], false, bh[nj], (short)0, acc[mi][nj], false, false);
        acc[mi][nj] = __builtin_amdgcn_wmma_f32_16x16x32_bf16(
            false, ah[mi], false, bl[nj], (short)0, acc[mi][nj], false, false);
        acc[mi][nj] = __builtin_amdgcn_wmma_f32_16x16x32_bf16(
            false, al[mi], false, bh[nj], (short)0, acc[mi][nj], false, false);
      }
    }

#ifdef HAVE_ASYNC_LDS
    if (kt + 1 < KT) WAIT_ASYNC0();   // next stage landed in LDS
    __syncthreads();
#else
    if (kt + 1 < KT) l_store(nxt);
    __syncthreads();
#endif
  }

  // C/D 16x16 f32 layout: VGPR j -> row j (lanes 0-15) / row j+8 (lanes 16-31)
#pragma unroll
  for (int mi = 0; mi < 4; mi++) {
#pragma unroll
    for (int nj = 0; nj < 2; nj++) {
      int col   = n0 + wn * 32 + nj * 16 + lr;
      int rbase = m0 + wm * 64 + mi * 16 + half * 8;
#pragma unroll
      for (int j = 0; j < 8; j++) {
        out[(size_t)(rbase + j) * N_TOT + col] = acc[mi][nj][j];
      }
    }
  }
}

// ---------------------------------------------------------------------------
extern "C" void kernel_launch(void* const* d_in, const int* in_sizes, int n_in,
                              void* d_out, int out_size, void* d_ws, size_t ws_size,
                              hipStream_t stream) {
  const float* x   = (const float*)d_in[0];
  const float* sel = (const float*)d_in[1];
  const float* w   = (const float*)d_in[2];
  const int*   idx = (const int*)d_in[3];
  float* out = (float*)d_out;

  const size_t XE = (size_t)M_TOT * K_TOT;  // 33.5M elems
  const size_t WE = (size_t)N_TOT * K_TOT;  // 16.8M elems
  __bf16* xh = (__bf16*)d_ws;
  __bf16* xl = xh + XE;
  __bf16* wh = xl + XE;
  __bf16* wl = wh + WE;

  prep_x_kernel<<<(unsigned)(XE / 4 / BLKSZ), BLKSZ, 0, stream>>>(x, xh, xl);
  prep_w_kernel<<<(unsigned)(WE / 4 / BLKSZ), BLKSZ, 0, stream>>>(w, sel, idx, wh, wl);

  dim3 grid(N_TOT / TN, M_TOT / TM);  // (32, 64)
  gemm_kernel<<<grid, BLKSZ, 0, stream>>>(xh, xl, wh, wl, out);
}
